// QuantLinearDequant_59751585022726
// MI455X (gfx1250) — matrix operand order repaired and checked by reading, not compiled, not run on hardware
//
#include <hip/hip_runtime.h>

// ---------------------------------------------------------------------------
// Quantized linear: Y[m, o] = (sum_k x[m,k] * Wq[o,k]) * row_scales[o] + bias[o]
// M = 512, K = 4096, N = 16384.  fp32 out.
//
// MI455X plan:
//   - f16 WMMA (v_wmma_f32_16x16x32_f16), f32 accum. Wq in [-127,127] is exact
//     in f16; x ~ N(0,1) converts with ~5e-4 rel err.
//   - Each block owns a 64-wide N strip across ALL of M  => the 256 MB Wq
//     stream is read from HBM exactly once (roofline floor ~11 us @ 23.3 TB/s).
//   - W tile is staged through LDS: cooperatively loaded (int32) + dequantized
//     to f16 ONCE per block per K-stage (vs once per wave), double-buffered,
//     consumed via bank-conflict-free ds_load_b128.
// ---------------------------------------------------------------------------

typedef __attribute__((ext_vector_type(16))) _Float16 v16h;
typedef __attribute__((ext_vector_type(8)))  float    v8f;

#define IN_F   4096
#define OUT_F  16384
#define M_TOT  512
#define NTILE  64                    // N columns per block
#define KSTAGE 64                    // K per pipeline stage (2 WMMA k-steps)
#define NSTAGES (IN_F / KSTAGE)      // 64
#define LDS_STRIDE 72                // halves per LDS row: 64 + 8 pad (bank-safe)

// ---------------------------------------------------------------------------
// Kernel 1: x f32 -> f16 (float4 in, 4 halves out)
// ---------------------------------------------------------------------------
__global__ void __launch_bounds__(256)
cvt_x_to_f16(const float* __restrict__ x, _Float16* __restrict__ xh) {
    const int i = (blockIdx.x * 256 + threadIdx.x) * 4;
    float4 v = *(const float4*)(x + i);
    union { _Float16 h[4]; unsigned long long u; } r;
    r.h[0] = (_Float16)v.x;
    r.h[1] = (_Float16)v.y;
    r.h[2] = (_Float16)v.z;
    r.h[3] = (_Float16)v.w;
    *(unsigned long long*)(xh + i) = r.u;
}

// 8 int32 -> 8 f16, one b128 LDS store
__device__ __forceinline__ void cvt8_store(_Float16* dst, int4 a, int4 b) {
    union { _Float16 h[8]; int4 q; } r;
    r.h[0] = (_Float16)a.x;  r.h[1] = (_Float16)a.y;
    r.h[2] = (_Float16)a.z;  r.h[3] = (_Float16)a.w;
    r.h[4] = (_Float16)b.x;  r.h[5] = (_Float16)b.y;
    r.h[6] = (_Float16)b.z;  r.h[7] = (_Float16)b.w;
    *(int4*)dst = r.q;
}

// ---------------------------------------------------------------------------
// Kernel 2: WMMA GEMM, LDS-staged dequant, double-buffered K pipeline.
//
// Block: 512 threads = 16 waves. Block tile: M = 512 (all), N = 64.
// Wave w owns M rows [32w, 32w+32) = 2 subtiles of 16; 4 N subtiles of 16.
//
// Operand layouts per CDNA5 ISA 7.12.2 (16-bit, wave32):
//   A 16x32 : lane = row M = lane%16; halves 0..7 at k0+8*(lane>=16), 8..15 at +16
//   B 32x16 : lane = col N = lane%16; halves 0..7 at k0+8*(lane>=16), 8..15 at +16
//             -> two ds_load_b128 from the staged f16 tile
//   C/D f32 : VGPR r, lane L -> M = r + 8*(L/16), N = L%16
// ---------------------------------------------------------------------------
__global__ void __launch_bounds__(512)
qgemm_wmma_f16(const _Float16* __restrict__ xh,
               const int*      __restrict__ Wq,
               const float*    __restrict__ row_scales,
               const float*    __restrict__ bias,
               float*          __restrict__ out) {
    __shared__ _Float16 wtile[2][NTILE][LDS_STRIDE];   // 18 KB, double buffered

    const int tid    = threadIdx.x;
    const int lane   = tid & 31;
    const int wave   = tid >> 5;            // 0..15
    const int l16    = lane & 15;
    const int half16 = lane >> 4;           // 0 or 1
    const int n_base = blockIdx.x * NTILE;
    const int m_wave = wave * 32;

    // ---- cooperative W loader mapping: 64 cols x 8 k-chunks of 8 ints ----
    const int lcol = tid >> 3;              // 0..63
    const int lkp  = tid & 7;               // 0..7
    const int* wload = Wq + (size_t)(n_base + lcol) * IN_F + lkp * 8;

    // ---- accumulators: [mt][nt] = 2 x 4 tiles of 16x16 -> 64 VGPRs ----
    v8f acc[2][4];
#pragma unroll
    for (int i = 0; i < 2; ++i)
#pragma unroll
        for (int j = 0; j < 4; ++j)
#pragma unroll
            for (int r = 0; r < 8; ++r)
                acc[i][j][r] = 0.0f;

    const _Float16* abase[2];
#pragma unroll
    for (int mt = 0; mt < 2; ++mt)
        abase[mt] = xh + (size_t)(m_wave + mt * 16 + l16) * IN_F + half16 * 8;

    // ---- prologue: stage 0 into buffer 0 ----
    {
        int4 wa = *(const int4*)(wload);
        int4 wb = *(const int4*)(wload + 4);
        cvt8_store(&wtile[0][lcol][lkp * 8], wa, wb);
    }
    __syncthreads();

    // ---- main pipeline over 64 K-stages of 64 ----
    for (int s = 0; s < NSTAGES; ++s) {
        const int k0 = s * KSTAGE;

        // 1) issue next stage's global W loads early (latency hidden by WMMA)
        int4 wa = {}, wb = {};
        if (s + 1 < NSTAGES) {
            const int* wp = wload + (k0 + KSTAGE);
            wa = *(const int4*)(wp);
            wb = *(const int4*)(wp + 4);
            if (s + 2 < NSTAGES)
                __builtin_prefetch((const void*)(wp + KSTAGE), 0, 1);
        }

        // 2) compute stage s from LDS buffer (s & 1)
        const _Float16* lbuf = &wtile[s & 1][0][0];
#pragma unroll
        for (int ks = 0; ks < 2; ++ks) {
            v16h b[4];
#pragma unroll
            for (int nt = 0; nt < 4; ++nt) {
                const _Float16* lp = lbuf + (nt * 16 + l16) * LDS_STRIDE
                                          + ks * 32 + half16 * 8;
                union { int4 q[2]; v16h h; } bv;
                bv.q[0] = *(const int4*)(lp);        // ds_load_b128
                bv.q[1] = *(const int4*)(lp + 16);   // ds_load_b128
                b[nt] = bv.h;
            }
#pragma unroll
            for (int mt = 0; mt < 2; ++mt) {
                const _Float16* ap = abase[mt] + k0 + ks * 32;
                union { int4 q[2]; v16h h; } av;
                av.q[0] = *(const int4*)(ap);        // global b128
                av.q[1] = *(const int4*)(ap + 16);   // global b128
                const v16h a = av.h;
#pragma unroll
                for (int nt = 0; nt < 4; ++nt) {
                    acc[mt][nt] = __builtin_amdgcn_wmma_f32_16x16x32_f16(
                        false, a, false, b[nt],
                        (short)0, acc[mt][nt], false, false);
                }
            }
        }

        // 3) dequant + store next stage into the other buffer
        if (s + 1 < NSTAGES)
            cvt8_store(&wtile[(s + 1) & 1][lcol][lkp * 8], wa, wb);

        __syncthreads();   // next-buffer stores visible; safe to overwrite later
    }

    // ---- epilogue: per-column scale + bias ----
#pragma unroll
    for (int nt = 0; nt < 4; ++nt) {
        const int col = n_base + nt * 16 + l16;
        const float sc = row_scales[col];
        const float bz = bias[col];
#pragma unroll
        for (int mt = 0; mt < 2; ++mt) {
            const int row0 = m_wave + mt * 16 + half16 * 8;
#pragma unroll
            for (int r = 0; r < 8; ++r) {
                out[(size_t)(row0 + r) * OUT_F + col] = acc[mt][nt][r] * sc + bz;
            }
        }
    }
}

// ---------------------------------------------------------------------------
// Launch
// ---------------------------------------------------------------------------
extern "C" void kernel_launch(void* const* d_in, const int* in_sizes, int n_in,
                              void* d_out, int out_size, void* d_ws, size_t ws_size,
                              hipStream_t stream) {
    const float* x          = (const float*)d_in[0];   // [8,64,4096] f32
    const int*   Wq         = (const int*)  d_in[1];   // [16384,4096] int32
    const float* row_scales = (const float*)d_in[2];   // [16384] f32
    const float* bias       = (const float*)d_in[3];   // [16384] f32
    float*       out        = (float*)d_out;           // [8,64,16384] f32

    _Float16* xh = (_Float16*)d_ws;                    // 4 MB scratch

    const int n_cvt_blocks = (M_TOT * IN_F / 4) / 256; // 2048, exact
    cvt_x_to_f16<<<n_cvt_blocks, 256, 0, stream>>>(x, xh);

    qgemm_wmma_f16<<<OUT_F / NTILE, 512, 0, stream>>>(xh, Wq, row_scales, bias, out);
}
